// LSTMCell_6116033429913
// MI455X (gfx1250) — matrix operand order repaired and checked by reading, not compiled, not run on hardware
//
#include <hip/hip_runtime.h>
#include <stdint.h>

// ---------------------------------------------------------------------------
// Fused LSTM cell for MI455X (gfx1250, wave32, WMMA, async global->LDS).
// Pass 1: fp32 -> bf16 conversion of [x_t|h_prev] and gate-major W4 into ws.
// Pass 2: bf16 WMMA GEMM (gates = A @ W4^T) with ASYNCcnt-tracked
//         global_load_async_to_lds_b128 staging, double-buffered LDS, and a
//         fully fused in-register sigmoid/tanh LSTM epilogue.
// ---------------------------------------------------------------------------

typedef __attribute__((ext_vector_type(16))) __bf16 v16bf;
typedef __attribute__((ext_vector_type(8)))  __bf16 v8bf;
typedef __attribute__((ext_vector_type(8)))  float  v8f;
typedef __attribute__((ext_vector_type(4)))  float  v4f;
typedef __attribute__((ext_vector_type(4)))  int    v4i;

#define KTOT  2048     // INPUT + HIDDEN
#define HID   1024
#define BATCH 8192
#define BM    256      // block M tile
#define BK    32       // K step (one wmma K-depth)
#define LDA   40       // padded bf16 row stride (80B -> bank-conflict-free)
#define LDB   40

#if defined(__has_builtin)
#if __has_builtin(__builtin_amdgcn_global_load_async_to_lds_b128)
#define HAVE_ASYNC_LDS 1
#endif
#if __has_builtin(__builtin_amdgcn_s_wait_asynccnt)
#define HAVE_WAIT_ASYNC 1
#endif
#endif

// Builtin signature (from hipcc diagnostic): param0 is addrspace(1) v4i*.
typedef __attribute__((address_space(1))) v4i gv4i_t;
typedef __attribute__((address_space(3))) v4i lv4i_t;

// 16B global -> LDS copy. Async (ASYNCcnt, no VGPR data) when available.
__device__ __forceinline__ void copy16_g2l(const __bf16* gsrc, __bf16* ldst) {
#ifdef HAVE_ASYNC_LDS
    __builtin_amdgcn_global_load_async_to_lds_b128(
        (gv4i_t*)(uintptr_t)gsrc,
        (lv4i_t*)(uint32_t)(uintptr_t)ldst, /*offset=*/0, /*cpol=*/0);
#else
    *(v8bf*)ldst = *(const v8bf*)gsrc;
#endif
}

__device__ __forceinline__ void wait_async_all() {
#ifdef HAVE_ASYNC_LDS
#ifdef HAVE_WAIT_ASYNC
    __builtin_amdgcn_s_wait_asynccnt(0);
#else
    asm volatile("s_wait_asynccnt 0" ::: "memory");
#endif
#endif
}

__device__ __forceinline__ float fast_sigmoid(float x) {
    return 1.0f / (1.0f + __expf(-x));
}
__device__ __forceinline__ float fast_tanh(float x) {
    float e = __expf(-2.0f * x);
    return (1.0f - e) / (1.0f + e);
}

// ---------------- pass 1: fp32 -> bf16 staging kernels ----------------

__global__ __launch_bounds__(256)
void convert_A(const float* __restrict__ x_t, const float* __restrict__ h_prev,
               __bf16* __restrict__ out) {
    unsigned i   = (blockIdx.x * 256u + threadIdx.x) * 8u;  // over 8192*2048
    unsigned row = i >> 11;
    unsigned col = i & 2047u;
    const float* src = (col < HID) ? (x_t + row * HID + col)
                                   : (h_prev + row * HID + (col - HID));
    v4f a = *(const v4f*)src;
    v4f b = *(const v4f*)(src + 4);
    v8bf o = {(__bf16)a.x, (__bf16)a.y, (__bf16)a.z, (__bf16)a.w,
              (__bf16)b.x, (__bf16)b.y, (__bf16)b.z, (__bf16)b.w};
    *(v8bf*)(out + i) = o;
}

__global__ __launch_bounds__(256)
void convert_W(const float* __restrict__ Wf, const float* __restrict__ Wi,
               const float* __restrict__ Wc, const float* __restrict__ Wo,
               __bf16* __restrict__ out) {
    const float* Wg[4] = {Wf, Wi, Wc, Wo};
    unsigned gate = blockIdx.y;
    unsigned i    = (blockIdx.x * 256u + threadIdx.x) * 8u;  // over 1024*2048
    const float* src = Wg[gate] + i;
    v4f a = *(const v4f*)src;
    v4f b = *(const v4f*)(src + 4);
    v8bf o = {(__bf16)a.x, (__bf16)a.y, (__bf16)a.z, (__bf16)a.w,
              (__bf16)b.x, (__bf16)b.y, (__bf16)b.z, (__bf16)b.w};
    *(v8bf*)(out + (size_t)gate * HID * KTOT + i) = o;
}

// ---------------- pass 2: WMMA GEMM + fused LSTM epilogue ----------------

__global__ __launch_bounds__(256)
void lstm_gemm_fused(const __bf16* __restrict__ Abf,   // [8192][2048]
                     const __bf16* __restrict__ Bbf,   // [4][1024][2048]
                     const float*  __restrict__ c_prev,
                     const float* __restrict__ bF, const float* __restrict__ bI,
                     const float* __restrict__ bC, const float* __restrict__ bO,
                     float* __restrict__ h_out, float* __restrict__ c_out)
{
    __shared__ __bf16 As[2][BM * LDA];   // 2*256*40*2B = 40 KB
    __shared__ __bf16 Bs[2][128 * LDB];  // 2*128*40*2B = 20 KB

    const int t     = threadIdx.x;
    const int wave  = t >> 5;
    const int lane  = t & 31;
    const int l16   = lane & 15;
    const int lhalf = lane >> 4;

    const int m0 = blockIdx.y * BM;   // batch-row origin
    const int n0 = blockIdx.x * 32;   // hidden-column origin (per gate)

    v8f acc[2][8] = {};               // 16 x (16x16 f32) accumulators / wave

    // 6 x 16B async copies per thread per K-step (A: 4, B: 2)
    auto stage = [&](int k0, int buf) {
#pragma unroll
        for (int it = 0; it < 4; ++it) {          // A tile: 256 rows x 64B
            int id  = it * 256 + t;
            int row = id >> 2;
            int c   = (id & 3) * 8;
            copy16_g2l(Abf + (unsigned)(m0 + row) * KTOT + k0 + c,
                       &As[buf][row * LDA + c]);
        }
#pragma unroll
        for (int it = 0; it < 2; ++it) {          // B tile: 128 rows x 64B
            int id   = it * 256 + t;
            int row  = id >> 2;                   // 0..127
            int c    = (id & 3) * 8;
            int gate = row >> 5;
            int ncol = row & 31;
            copy16_g2l(Bbf + ((unsigned)gate * HID + n0 + ncol) * KTOT + k0 + c,
                       &Bs[buf][row * LDB + c]);
        }
    };

    auto compute = [&](int buf) {
        // A frag (16-bit 16x32 layout): lanes 0-15 K{0..7,16..23} of row l16,
        // lanes 16-31 K{8..15,24..31} -> two 16B LDS chunks per lane.
        v16bf afrag[2];
#pragma unroll
        for (int mt = 0; mt < 2; ++mt) {
            int row = wave * 32 + mt * 16 + l16;
            union { v16bf v; v8bf h[2]; } u;
            u.h[0] = *(const v8bf*)&As[buf][row * LDA + lhalf * 8];
            u.h[1] = *(const v8bf*)&As[buf][row * LDA + 16 + lhalf * 8];
            afrag[mt] = u.v;
        }
#pragma unroll
        for (int j = 0; j < 8; ++j) {
            // B frag: lane = column j*16+l16, K{0..7,16..23} (lanes 0-15) / K{8..15,24..31} (lanes 16-31)
            int brow = j * 16 + l16;
            union { v16bf v; v8bf h[2]; } ub;
            ub.h[0] = *(const v8bf*)&Bs[buf][brow * LDB + lhalf * 8];
            ub.h[1] = *(const v8bf*)&Bs[buf][brow * LDB + 16 + lhalf * 8];
            v16bf bfrag = ub.v;
#pragma unroll
            for (int mt = 0; mt < 2; ++mt) {
                acc[mt][j] = __builtin_amdgcn_wmma_f32_16x16x32_bf16(
                    false, afrag[mt], false, bfrag,
                    (short)0, acc[mt][j], false, false);
            }
        }
    };

    stage(0, 0);
    wait_async_all();
    __syncthreads();
#pragma unroll 1
    for (int kk = 0; kk < KTOT / BK; ++kk) {
        int cur = kk & 1;
        if (kk + 1 < KTOT / BK) stage((kk + 1) * BK, cur ^ 1);
        compute(cur);
        wait_async_all();   // drain this wave's async writes, then publish
        __syncthreads();
    }

    // fused LSTM epilogue: tiles j {0,1}=f {2,3}=i {4,5}=g {6,7}=o
    // C/D layout: lanes 0-15 M=r, lanes 16-31 M=8+r, N=l16.
#pragma unroll
    for (int mt = 0; mt < 2; ++mt) {
#pragma unroll
        for (int jj = 0; jj < 2; ++jj) {
            int col = n0 + jj * 16 + l16;
            float vbf = bF[col], vbi = bI[col], vbc = bC[col], vbo = bO[col];
            v8f fA = acc[mt][0 + jj];
            v8f iA = acc[mt][2 + jj];
            v8f gA = acc[mt][4 + jj];
            v8f oA = acc[mt][6 + jj];
#pragma unroll
            for (int r = 0; r < 8; ++r) {
                int m = m0 + wave * 32 + mt * 16 + lhalf * 8 + r;
                float f = fast_sigmoid(fA[r] + vbf);
                float i = fast_sigmoid(iA[r] + vbi);
                float g = fast_tanh  (gA[r] + vbc);
                float o = fast_sigmoid(oA[r] + vbo);
                unsigned idx = (unsigned)m * HID + col;
                float ct = f * c_prev[idx] + i * g;
                float ht = o * fast_tanh(ct);
                h_out[idx] = ht;
                c_out[idx] = ct;
            }
        }
    }
}

// ---------------------------------------------------------------------------

extern "C" void kernel_launch(void* const* d_in, const int* in_sizes, int n_in,
                              void* d_out, int out_size, void* d_ws, size_t ws_size,
                              hipStream_t stream) {
    const float* x_t    = (const float*)d_in[0];
    const float* h_prev = (const float*)d_in[1];
    const float* c_prev = (const float*)d_in[2];
    const float* Wf     = (const float*)d_in[3];
    const float* bF     = (const float*)d_in[4];
    const float* Wi     = (const float*)d_in[5];
    const float* bI     = (const float*)d_in[6];
    const float* Wc     = (const float*)d_in[7];
    const float* bC     = (const float*)d_in[8];
    const float* Wo     = (const float*)d_in[9];
    const float* bO     = (const float*)d_in[10];

    __bf16* Abf = (__bf16*)d_ws;                       // 32 MB
    __bf16* Bbf = Abf + (size_t)BATCH * KTOT;          // 16 MB

    float* h_out = (float*)d_out;
    float* c_out = h_out + (size_t)BATCH * HID;

    // pass 1: fp32 -> bf16 (one-shot; ~2us of HBM traffic)
    convert_A<<<(BATCH * KTOT) / (8 * 256), 256, 0, stream>>>(x_t, h_prev, Abf);
    convert_W<<<dim3((HID * KTOT) / (8 * 256), 4), 256, 0, stream>>>(Wf, Wi, Wc, Wo, Bbf);

    // pass 2: fused WMMA GEMM + LSTM pointwise
    dim3 grid(HID / 32, BATCH / BM);   // 32 x 32 blocks
    lstm_gemm_fused<<<grid, 256, 0, stream>>>(Abf, Bbf, c_prev,
                                              bF, bI, bC, bO, h_out, c_out);
}
